// SelfAttentionConv_54546084659812
// MI455X (gfx1250) — matrix operand order repaired and checked by reading, not compiled, not run on hardware
//
#include <hip/hip_runtime.h>

// ---------------------------------------------------------------------------
// MI455X (gfx1250) implementation.
// Compute-bound (~25 GFLOP, <60MB working set -> resident in 192MB L2;
// HBM 23.3 TB/s is not the limiter). All matmuls use v_wmma_f32_16x16x32_f16
// (f16 in, f32 accumulate), wave32 layouts per CDNA5 ISA.
// Causal masking is hoisted out of the steady-state flash loop: only the
// final s-chunk of each q-tile needs mask VALU work.
// ---------------------------------------------------------------------------

typedef __attribute__((ext_vector_type(16))) _Float16 v16h;
typedef __attribute__((ext_vector_type(8)))  _Float16 v8h;
typedef __attribute__((ext_vector_type(8)))  float    v8f;

#define B_  8
#define T_  1024
#define K_  64
#define H_  8
#define WIN 5

__device__ __forceinline__ float silu_f(float v) { return v / (1.f + __expf(-v)); }

__device__ __forceinline__ v8f wmma16(v16h a, v16h b, v8f c) {
  // D = A(16x32 f16) * B(32x16 f16) + C(16x16 f32)
  return __builtin_amdgcn_wmma_f32_16x16x32_f16(
      /*neg_a=*/false, a, /*neg_b=*/false, b,
      /*c_mod=*/(short)0, c, /*reuse_a=*/false, /*reuse_b=*/false);
}

// A operand: 16x32 f16 tile from row-major [rows][64] matrix, K-chunk dbase.
// lanes 0-15: e0..15 -> K = dbase+0..15
// lanes16-31: e0..15 -> K = dbase+16..31
__device__ __forceinline__ v16h loadA16(const _Float16* base, int row, int dbase,
                                        int hf) {
  const _Float16* p = base + (long)row * K_ + dbase + hf * 16;
  v8h lo = *(const v8h*)p;
  v8h hi = *(const v8h*)(p + 8);
  v16h a;
#pragma unroll
  for (int i = 0; i < 8; ++i) { a[i] = lo[i]; a[i + 8] = hi[i]; }
  return a;
}

// B operand for scores: B = K^T (d x s). Source is K stored row-major [s][64].
// lane: n = scol+lane16 ; elements e -> K-dim d = dbase + 16*hf + e (contig).
__device__ __forceinline__ v16h loadB_k(const _Float16* base, int scol, int dbase,
                                        int lane16, int hf) {
  const _Float16* p = base + (long)(scol + lane16) * K_ + dbase + hf * 16;
  return *(const v16h*)p;
}

// B operand for PV: B = V (s x d). Source is V^T stored [64][1024] (d-major),
// so per-lane elements (varying s) are contiguous.
// lane: n = dt*16+lane16 ; elements e -> K-dim s = s0 + 16*hf + e.
__device__ __forceinline__ v16h loadB_v(const _Float16* baseT, int dt, int s0,
                                        int lane16, int hf) {
  const _Float16* p = baseT + (long)(dt * 16 + lane16) * T_ + s0 + hf * 16;
  return *(const v16h*)p;
}

// ---------------------------------------------------------------------------
// Kernel 1: dilated + 1x1 convs with SiLU, written directly in the permuted
// "flat" attention layout.  n = b*8+j, q mixed index: t = j*128+q/8, h = q%8,
// conv channel c = d*8 + h  (reshape (b, k, h, t)).
// ---------------------------------------------------------------------------
__global__ void conv_silu_kernel(const float* __restrict__ x,
                                 const float* __restrict__ Wq, const float* __restrict__ bq,
                                 const float* __restrict__ Wk, const float* __restrict__ bk,
                                 const float* __restrict__ Wv, const float* __restrict__ Wl,
                                 _Float16* __restrict__ Qg, _Float16* __restrict__ Kg,
                                 _Float16* __restrict__ Vt, _Float16* __restrict__ Lq,
                                 _Float16* __restrict__ Lt) {
  int nq = blockIdx.x;
  int n = nq >> 10, q = nq & 1023;
  int b = n >> 3, j = n & 7;
  int t = j * 128 + (q >> 3);
  int hh = q & 7;
  int d = threadIdx.x;      // 0..63
  int c = d * 8 + hh;       // output channel in (k,h) split

  __shared__ float xs[5][K_];
#pragma unroll
  for (int s = 0; s < 5; ++s) {
    int tt = t - 8 + 2 * s;               // dilation-2, left pad 8
    xs[s][d] = (tt >= 0) ? x[((long)b * T_ + tt) * K_ + d] : 0.f;
  }
  __syncthreads();

  float accq = bq[c], acck = bk[c], accv = 0.f, accl = 0.f;
  const float* wv = Wv + (long)c * K_;
  const float* wl = Wl + (long)c * K_;
#pragma unroll 8
  for (int ci = 0; ci < K_; ++ci) {
    float xv = xs[4][ci];                 // row t (1x1 convs)
    accv = fmaf(xv, wv[ci], accv);
    accl = fmaf(xv, wl[ci], accl);
  }
  const float* wq = Wq + (long)c * K_ * 5;
  const float* wk = Wk + (long)c * K_ * 5;
#pragma unroll
  for (int s = 0; s < 5; ++s) {
#pragma unroll 8
    for (int ci = 0; ci < K_; ++ci) {
      float xv = xs[s][ci];
      accq = fmaf(xv, wq[ci * 5 + s], accq);
      acck = fmaf(xv, wk[ci * 5 + s], acck);
    }
  }
  float vq = silu_f(accq), vk = silu_f(acck);
  float vv = silu_f(accv), vl = silu_f(accl);
  long rowi = ((long)n * T_ + q) * K_ + d;   // [n][q][d]
  long coli = ((long)n * K_ + d) * T_ + q;   // [n][d][q] (transposed)
  Qg[rowi] = (_Float16)vq;
  Kg[rowi] = (_Float16)vk;
  Lq[rowi] = (_Float16)vl;
  Vt[coli] = (_Float16)vv;
  Lt[coli] = (_Float16)vl;
}

// ---------------------------------------------------------------------------
// One 32-wide flash-attention s-chunk.
//   MODE 0: no masking (steady-state, s0+31 <= q0)
//   MODE 1: causal tail mask (s <= q)
//   MODE 2: windowed local mask (0 <= s <= q, q-s <= WIN)
// ---------------------------------------------------------------------------
template <int MODE>
__device__ __forceinline__ void flash_chunk(
    const _Float16* __restrict__ Kmat, const _Float16* __restrict__ VmatT,
    v16h alo, v16h ahi, int s0, int q0, int lane16, int hf,
    float (&m)[8], float (&l)[8], v8f (&og)[4], _Float16 (*Plds)[32]) {
  const float SCALE = 0.125f;  // 1/sqrt(K)

  v16h b00 = loadB_k(Kmat, s0,      0,  lane16, hf);
  v16h b01 = loadB_k(Kmat, s0,      32, lane16, hf);
  v16h b10 = loadB_k(Kmat, s0 + 16, 0,  lane16, hf);
  v16h b11 = loadB_k(Kmat, s0 + 16, 32, lane16, hf);
  v8f z = {};
  v8f S0 = wmma16(alo, b00, z); S0 = wmma16(ahi, b01, S0);
  v8f S1 = wmma16(alo, b10, z); S1 = wmma16(ahi, b11, S1);

  float p0[8], p1[8];
  int sc0 = s0 + lane16, sc1 = s0 + 16 + lane16;
#pragma unroll
  for (int i = 0; i < 8; ++i) {
    float v0 = S0[i] * SCALE;
    float v1 = S1[i] * SCALE;
    if (MODE != 0) {
      int qrow = q0 + hf * 8 + i;
      bool ok0 = (sc0 <= qrow), ok1 = (sc1 <= qrow);
      if (MODE == 2) {
        ok0 = ok0 && (sc0 >= 0) && (qrow - sc0 <= WIN);
        ok1 = ok1 && (sc1 >= 0) && (qrow - sc1 <= WIN);
      }
      v0 = ok0 ? v0 : -1e30f;
      v1 = ok1 ? v1 : -1e30f;
    }
    float mx = fmaxf(v0, v1);
    for (int off = 1; off < 16; off <<= 1) mx = fmaxf(mx, __shfl_xor(mx, off));
    float mnew = fmaxf(m[i], mx);
    float e0 = __expf(v0 - mnew), e1 = __expf(v1 - mnew);
    float sm = e0 + e1;
    for (int off = 1; off < 16; off <<= 1) sm += __shfl_xor(sm, off);
    float corr = __expf(m[i] - mnew);
    l[i] = l[i] * corr + sm;
    m[i] = mnew;
#pragma unroll
    for (int dt = 0; dt < 4; ++dt) og[dt][i] *= corr;
    p0[i] = e0; p1[i] = e1;
  }

  __syncthreads();
#pragma unroll
  for (int i = 0; i < 8; ++i) {
    Plds[hf * 8 + i][lane16]      = (_Float16)p0[i];
    Plds[hf * 8 + i][16 + lane16] = (_Float16)p1[i];
  }
  __syncthreads();
  v8h plo = *(const v8h*)&Plds[lane16][hf * 16];
  v8h phi = *(const v8h*)&Plds[lane16][hf * 16 + 8];
  v16h Pa;
#pragma unroll
  for (int i = 0; i < 8; ++i) { Pa[i] = plo[i]; Pa[i + 8] = phi[i]; }
#pragma unroll
  for (int dt = 0; dt < 4; ++dt)
    og[dt] = wmma16(Pa, loadB_v(VmatT, dt, s0, lane16, hf), og[dt]);
}

// ---------------------------------------------------------------------------
// Kernel 2: fused local(window<=5) + global causal attention + gating.
// One wave per (head n, 16-row q tile). WMMA flash-attention.
// ---------------------------------------------------------------------------
__global__ __launch_bounds__(32)
void attention_kernel(const _Float16* __restrict__ Qg, const _Float16* __restrict__ Kg,
                      const _Float16* __restrict__ Vt, const _Float16* __restrict__ Lq,
                      const _Float16* __restrict__ Lt, _Float16* __restrict__ G) {
  int bx = blockIdx.x;
  int n = bx >> 6;
  int q0 = (bx & 63) * 16;
  int L = threadIdx.x;
  int hf = L >> 4, lane16 = L & 15;
  int bb = n >> 3, j = n & 7;

  const _Float16* Qn  = Qg + (long)n * T_ * K_;
  const _Float16* Kn  = Kg + (long)n * T_ * K_;
  const _Float16* Vn  = Vt + (long)n * K_ * T_;
  const _Float16* Ln  = Lq + (long)n * T_ * K_;
  const _Float16* LTn = Lt + (long)n * K_ * T_;

  __shared__ __align__(32) _Float16 Plds[16][32];

  // -------------------- local (windowed) branch: single 32-chunk ----------
  v8f ol[4] = {};
  float ml[8], ll[8];
#pragma unroll
  for (int i = 0; i < 8; ++i) { ml[i] = -1e30f; ll[i] = 0.f; }
  {
    v16h alo = loadA16(Ln, q0 + lane16, 0, hf);
    v16h ahi = loadA16(Ln, q0 + lane16, 32, hf);
    // chunk [q0-16, q0+15] covers the window [q-5, q] for all 16 rows
    flash_chunk<2>(Ln, LTn, alo, ahi, q0 - 16, q0, lane16, hf, ml, ll, ol, Plds);
#pragma unroll
    for (int i = 0; i < 8; ++i) {
      float inv = 1.f / ll[i];
#pragma unroll
      for (int dt = 0; dt < 4; ++dt) ol[dt][i] *= inv;
    }
  }

  // -------------------- global (dense causal) branch: flash loop ----------
  v8f og[4] = {};
  float m[8], l[8];
#pragma unroll
  for (int i = 0; i < 8; ++i) { m[i] = -1e30f; l[i] = 0.f; }
  {
    v16h alo = loadA16(Qn, q0 + lane16, 0, hf);
    v16h ahi = loadA16(Qn, q0 + lane16, 32, hf);
    // chunks with s0+31 <= q0 are fully inside the causal region: no masking.
    int s0_mask = (q0 + 15) & ~31;     // the single chunk needing the mask
    for (int s0 = 0; s0 < s0_mask; s0 += 32)
      flash_chunk<0>(Kn, Vn, alo, ahi, s0, q0, lane16, hf, m, l, og, Plds);
    flash_chunk<1>(Kn, Vn, alo, ahi, s0_mask, q0, lane16, hf, m, l, og, Plds);
#pragma unroll
    for (int i = 0; i < 8; ++i) {
      float inv = 1.f / l[i];
#pragma unroll
      for (int dt = 0; dt < 4; ++dt) og[dt][i] *= inv;
    }
  }

  // -------------------- gate + concat epilogue ----------------------------
#pragma unroll
  for (int dt = 0; dt < 4; ++dt) {
#pragma unroll
    for (int i = 0; i < 8; ++i) {
      int qrow = q0 + hf * 8 + i;
      int d = dt * 16 + lane16;
      float gv = og[dt][i];
      float lv = ol[dt][i];
      float gate = 1.f / (1.f + __expf(-gv));
      long base = ((long)bb * T_ + qrow) * (2 * H_ * K_);
      G[base + (long)j * K_ + d]        = (_Float16)((1.f - gate) * lv);
      G[base + (long)(H_ + j) * K_ + d] = (_Float16)(gate * gv);
    }
  }
}

// ---------------------------------------------------------------------------
// Kernel 3: Y = silu(G(8192x1024) * Wu^T(1024x64) + bu), WMMA.
// ---------------------------------------------------------------------------
__global__ __launch_bounds__(32)
void out_gemm_kernel(const _Float16* __restrict__ G, const _Float16* __restrict__ Wuh,
                     const float* __restrict__ bu, float* __restrict__ Y) {
  int r0 = blockIdx.x * 16;
  int L = threadIdx.x;
  int hf = L >> 4, lane16 = L & 15;
  v8f acc[4] = {};
  for (int k0 = 0; k0 < 2 * H_ * K_; k0 += 32) {
    const _Float16* pa = G + (long)(r0 + lane16) * (2 * H_ * K_) + k0 + hf * 16;
    v8h lo = *(const v8h*)pa;
    v8h hi = *(const v8h*)(pa + 8);
    v16h A;
#pragma unroll
    for (int i = 0; i < 8; ++i) { A[i] = lo[i]; A[i + 8] = hi[i]; }
#pragma unroll
    for (int nt = 0; nt < 4; ++nt) {
      const _Float16* pb = Wuh + (long)(nt * 16 + lane16) * (2 * H_ * K_) + k0 + hf * 16;
      v16h Bv = *(const v16h*)pb;
      acc[nt] = wmma16(A, Bv, acc[nt]);
    }
  }
#pragma unroll
  for (int nt = 0; nt < 4; ++nt) {
#pragma unroll
    for (int i = 0; i < 8; ++i) {
      int row = r0 + hf * 8 + i;
      int ncol = nt * 16 + lane16;
      float y = acc[nt][i] + bu[ncol];
      Y[(long)row * K_ + ncol] = silu_f(y);
    }
  }
}

__global__ void cvt_wu_kernel(const float* __restrict__ Wu, _Float16* __restrict__ Wuh) {
  int i = blockIdx.x * 256 + threadIdx.x;
  if (i < K_ * 2 * H_ * K_) Wuh[i] = (_Float16)Wu[i];
}

// ---------------------------------------------------------------------------
extern "C" void kernel_launch(void* const* d_in, const int* in_sizes, int n_in,
                              void* d_out, int out_size, void* d_ws, size_t ws_size,
                              hipStream_t stream) {
  const float* x  = (const float*)d_in[0];
  const float* Wq = (const float*)d_in[1];
  const float* bq = (const float*)d_in[2];
  const float* Wk = (const float*)d_in[3];
  const float* bk = (const float*)d_in[4];
  const float* Wv = (const float*)d_in[5];
  const float* Wl = (const float*)d_in[6];
  const float* Wu = (const float*)d_in[7];
  const float* bu = (const float*)d_in[8];
  float* Y = (float*)d_out;

  char* ws = (char*)d_ws;
  const size_t SZ = (size_t)64 * T_ * K_ * sizeof(_Float16);  // 8 MB per tensor
  _Float16* Qg  = (_Float16*)(ws);
  _Float16* Kg  = (_Float16*)(ws + SZ);
  _Float16* Vt  = (_Float16*)(ws + 2 * SZ);
  _Float16* Lq  = (_Float16*)(ws + 3 * SZ);
  _Float16* Lt  = (_Float16*)(ws + 4 * SZ);
  _Float16* G   = (_Float16*)(ws + 5 * SZ);                    // 16 MB
  _Float16* Wuh = (_Float16*)(ws + 5 * SZ + (size_t)B_ * T_ * 2 * H_ * K_ * sizeof(_Float16));

  conv_silu_kernel<<<64 * T_, K_, 0, stream>>>(x, Wq, bq, Wk, bk, Wv, Wl,
                                               Qg, Kg, Vt, Lq, Lt);
  cvt_wu_kernel<<<(K_ * 2 * H_ * K_) / 256, 256, 0, stream>>>(Wu, Wuh);
  attention_kernel<<<64 * 64, 32, 0, stream>>>(Qg, Kg, Vt, Lq, Lt, G);
  out_gemm_kernel<<<(B_ * T_) / 16, 32, 0, stream>>>(G, Wuh, bu, Y);
}
